// H2_PE_88098369176176
// MI455X (gfx1250) — matrix operand order repaired and checked by reading, not compiled, not run on hardware
//
#include <hip/hip_runtime.h>

// ---------------------------------------------------------------------------
// MI455X (gfx1250) implementation of the GNN forward pass.
//   - GEMMs use v_wmma_f32_16x16x32_f16 (f16 in, f32 accumulate).
//   - Weights are transposed+converted once per call to f16 [M, Kpad] so both
//     A and B WMMA fragments are contiguous 128-bit per-lane loads.
//   - segment_max uses monotonic-unsigned encoding + global_atomic_max_u32.
//   - segment_sum / softmax-denominator use global_atomic_add_f32 (L2-resident:
//     the 50000x256 feature matrix is 51MB < 192MB L2).
// Params flattening assumption: jax pytree sorted-key order ->
//   b1{gat0{W,al,ar,b},gat1{...},gin0{eps,mlp{W1,W2,b1,b2,bta,g}},gin1{...}},
//   b2{...}, h2_0{eps,mlp{...}}, h2_1{...}, out{W1,W2,b1,b2}  => 74 inputs.
// ---------------------------------------------------------------------------

#define NA    50000          // atoms (multiple of 16)
#define NAT   (NA / 16)      // 3125 row tiles
#define NCA   5000           // C2 nodes (actual)
#define NCP   5008           // C2 padded to multiple of 16
#define NCT   (NCP / 16)     // 313 row tiles
#define EB    400000
#define EI    80000
#define GRP   32

typedef __attribute__((ext_vector_type(16))) _Float16 v16h;
typedef __attribute__((ext_vector_type(8)))  _Float16 v8h;
typedef __attribute__((ext_vector_type(8)))  float    v8f;

// -------- monotonic float <-> unsigned encoding for atomic max ------------
__device__ __forceinline__ unsigned fenc(float x) {
  unsigned u = __float_as_uint(x);
  return (u & 0x80000000u) ? ~u : (u | 0x80000000u);
}
__device__ __forceinline__ float fdec(unsigned u) {
  unsigned v = (u & 0x80000000u) ? (u & 0x7fffffffu) : ~u;
  return __uint_as_float(v);
}
#define ENC_NEGINF 0x007fffffu   // fenc(-inf)

// ---------------------------- utility fills --------------------------------
__global__ void __launch_bounds__(256) fill_f32(float* p, float v, int n) {
  int t = blockIdx.x * 256 + threadIdx.x; if (t < n) p[t] = v;
}
__global__ void __launch_bounds__(256) fill_u32(unsigned* p, unsigned v, int n) {
  int t = blockIdx.x * 256 + threadIdx.x; if (t < n) p[t] = v;
}

// ------------------- weight transpose + f32->f16 convert -------------------
// W: [K, M] f32 row-major  ->  Wt: [M, Kpad] f16 row-major (zero padded K)
__global__ void __launch_bounds__(256) wt_prep(const float* __restrict__ W,
                                               _Float16* __restrict__ Wt,
                                               int K, int M, int Kpad, int total) {
  int t = blockIdx.x * 256 + threadIdx.x; if (t >= total) return;
  int m = t / Kpad, k = t - m * Kpad;
  Wt[t] = (k < K) ? (_Float16)W[(size_t)k * M + m] : (_Float16)0.f;
}

// ------------------------------- GEMM --------------------------------------
// C[Npad,256] = A[Npad,K](f16) @ B[K,256]  (Bt = B^T stored [256,K] f16)
// grid.x = Npad/16 row tiles; 8 waves, each wave owns 2 of 16 column tiles.
__global__ void __launch_bounds__(256) gemm_wmma(const _Float16* __restrict__ A,
                                                 const _Float16* __restrict__ Bt,
                                                 const float* __restrict__ bias,
                                                 float* __restrict__ C, int K) {
  const int M    = 256;
  const int lane = threadIdx.x & 31;
  const int wid  = threadIdx.x >> 5;
  const int hh   = lane >> 4;      // half-wave select
  const int l15  = lane & 15;
  const size_t arow = (size_t)(blockIdx.x * 16 + l15) * K + hh * 16;
  const int n0 = wid * 16 + l15;   // column tile wid
  const int n1 = n0 + 128;         // column tile wid+8
  const int kb = hh * 16;          // B K-group base within 32-step

  v8f c0 = {}; v8f c1 = {};
  for (int k0 = 0; k0 < K; k0 += 32) {
    // A fragment: contiguous k order, lane half hh holds k 16*hh..16*hh+15
    union { v16h v; v8h h[2]; } a;
    const _Float16* ap = A + arow + k0;
    a.h[0] = *(const v8h*)(ap);        // K = kbase..kbase+7
    a.h[1] = *(const v8h*)(ap + 8);    // K = kbase+8..kbase+15
    // B fragments: dense B 32x16, lanes 0-15 hold K 0..15, lanes 16-31 K 16..31
    const v16h b0 = *(const v16h*)(Bt + (size_t)n0 * K + k0 + kb);
    const v16h b1 = *(const v16h*)(Bt + (size_t)n1 * K + k0 + kb);
    c0 = __builtin_amdgcn_wmma_f32_16x16x32_f16(false, a.v, false, b0, (short)0, c0, false, false);
    c1 = __builtin_amdgcn_wmma_f32_16x16x32_f16(false, a.v, false, b1, (short)0, c1, false, false);
  }
  const int rbase = blockIdx.x * 16 + hh * 8;   // C/D layout: lane holds col l15
  const float bv0 = bias ? bias[n0] : 0.f;
  const float bv1 = bias ? bias[n1] : 0.f;
#pragma unroll
  for (int i = 0; i < 8; i++) {
    size_t r = (size_t)(rbase + i) * M;
    C[r + n0] = c0[i] + bv0;
    C[r + n1] = c1[i] + bv1;
  }
}

// ------------------------------ posenc -------------------------------------
__global__ void __launch_bounds__(256) posenc_kernel(const float* __restrict__ af,
                                                     _Float16* __restrict__ x16, int N) {
  int n = blockIdx.x * 256 + threadIdx.x; if (n >= N) return;
  float p0 = af[n * 4], p1 = af[n * 4 + 1], p2 = af[n * 4 + 2];
  _Float16* o = x16 + (size_t)n * 64;
  o[0] = (_Float16)p0; o[1] = (_Float16)p1; o[2] = (_Float16)p2;
  float f = 1.f;
  for (int fr = 0; fr < 10; fr++) {
    float v0 = p0 * f, v1 = p1 * f, v2 = p2 * f;
    o[3 + fr * 3 + 0]  = (_Float16)__sinf(v0);
    o[3 + fr * 3 + 1]  = (_Float16)__sinf(v1);
    o[3 + fr * 3 + 2]  = (_Float16)__sinf(v2);
    o[33 + fr * 3 + 0] = (_Float16)__cosf(v0);
    o[33 + fr * 3 + 1] = (_Float16)__cosf(v1);
    o[33 + fr * 3 + 2] = (_Float16)__cosf(v2);
    f *= 2.f;
  }
  o[63] = (_Float16)af[n * 4 + 3];
}

// ------------------------------- GAT ---------------------------------------
__global__ void __launch_bounds__(256) gat_elr(const float* __restrict__ F,
                                               const float* __restrict__ al,
                                               const float* __restrict__ ar,
                                               float* __restrict__ el,
                                               float* __restrict__ er, int total) {
  int t = blockIdx.x * 256 + threadIdx.x; if (t >= total) return;
  int n = t >> 2, h = t & 3;
  const float* fr = F + (size_t)n * 256 + h * 64;
  const float* a1 = al + h * 64;
  const float* a2 = ar + h * 64;
  float s1 = 0.f, s2 = 0.f;
  for (int d = 0; d < 64; d++) { float v = fr[d]; s1 += v * a1[d]; s2 += v * a2[d]; }
  el[t] = s1; er[t] = s2;
}

__global__ void __launch_bounds__(256) gat_edge1(const float* __restrict__ el,
                                                 const float* __restrict__ er,
                                                 const int* __restrict__ src,
                                                 const int* __restrict__ dst,
                                                 float* __restrict__ eE,
                                                 unsigned* __restrict__ mEnc, int total) {
  int t = blockIdx.x * 256 + threadIdx.x; if (t >= total) return;
  int e = t >> 2, h = t & 3;
  int s = src[e], d = dst[e];
  float v = el[s * 4 + h] + er[d * 4 + h];
  v = v > 0.f ? v : 0.2f * v;            // leaky_relu(0.2)
  eE[t] = v;
  atomicMax(&mEnc[d * 4 + h], fenc(v));  // edge-softmax max per dst
}

__global__ void __launch_bounds__(256) gat_edge2(const float* __restrict__ eE,
                                                 const int* __restrict__ dst,
                                                 const unsigned* __restrict__ mEnc,
                                                 float* __restrict__ aE,
                                                 float* __restrict__ ssum, int total) {
  int t = blockIdx.x * 256 + threadIdx.x; if (t >= total) return;
  int e = t >> 2, h = t & 3;
  int d = dst[e];
  float a = __expf(eE[t] - fdec(mEnc[d * 4 + h]));
  aE[t] = a;
  atomicAdd(&ssum[d * 4 + h], a);
}

__global__ void __launch_bounds__(256) gat_edge3(const float* __restrict__ aE,
                                                 const float* __restrict__ F,
                                                 const int* __restrict__ src,
                                                 const int* __restrict__ dst,
                                                 float* __restrict__ acc, int total) {
  int t = blockIdx.x * 256 + threadIdx.x; if (t >= total) return;
  int e = t >> 8, dd = t & 255, h = dd >> 6;
  atomicAdd(&acc[(size_t)dst[e] * 256 + dd], aE[e * 4 + h] * F[(size_t)src[e] * 256 + dd]);
}

__global__ void __launch_bounds__(256) gat_fin(const float* __restrict__ acc,
                                               const float* __restrict__ ssum,
                                               const float* __restrict__ bvec,
                                               float* __restrict__ H,
                                               _Float16* __restrict__ H16, int total) {
  int t = blockIdx.x * 256 + threadIdx.x; if (t >= total) return;
  int n = t >> 8, d = t & 255, h = d >> 6;
  float v = acc[t] / ssum[n * 4 + h] + bvec[d];
  v = v > 0.f ? v : 0.f;
  H[t] = v; H16[t] = (_Float16)v;
}

// ------------------------------- GIN ---------------------------------------
__global__ void __launch_bounds__(256) gin_init(const float* __restrict__ x,
                                                const float* __restrict__ epsp,
                                                float* __restrict__ acc, int total) {
  int t = blockIdx.x * 256 + threadIdx.x; if (t >= total) return;
  acc[t] = (1.f + epsp[0]) * x[t];
}

__global__ void __launch_bounds__(256) gin_scatter(const float* __restrict__ x,
                                                   const int* __restrict__ src,
                                                   const int* __restrict__ dst,
                                                   float* __restrict__ acc,
                                                   int Din, int total) {
  int t = blockIdx.x * 256 + threadIdx.x; if (t >= total) return;
  int e = t / Din, c = t - e * Din;
  atomicAdd(&acc[(size_t)dst[e] * Din + c], x[(size_t)src[e] * Din + c]);
}

// f32 [N,Din] -> f16 [N,Kpad] with zero K-padding
__global__ void __launch_bounds__(256) conv_pad_f16(const float* __restrict__ X,
                                                    _Float16* __restrict__ Y,
                                                    int Din, int Kpad, int total) {
  int t = blockIdx.x * 256 + threadIdx.x; if (t >= total) return;
  int n = t / Kpad, k = t - n * Kpad;
  Y[t] = (k < Din) ? (_Float16)X[(size_t)n * Din + k] : (_Float16)0.f;
}

// column-wise batchnorm statistics over the first Nact rows (256 columns)
__global__ void __launch_bounds__(256) bn_stats(const float* __restrict__ X,
                                                float* __restrict__ mu,
                                                float* __restrict__ var, int Nact) {
  int c = blockIdx.x, t = threadIdx.x;
  float s = 0.f, s2 = 0.f;
  for (int n = t; n < Nact; n += 256) {
    float v = X[(size_t)n * 256 + c]; s += v; s2 += v * v;
  }
  __shared__ float sh[256], sh2[256];
  sh[t] = s; sh2[t] = s2; __syncthreads();
  for (int o = 128; o > 0; o >>= 1) {
    if (t < o) { sh[t] += sh[t + o]; sh2[t] += sh2[t + o]; }
    __syncthreads();
  }
  if (t == 0) {
    float m = sh[0] / (float)Nact;
    mu[c] = m; var[c] = sh2[0] / (float)Nact - m * m;
  }
}

__global__ void __launch_bounds__(256) bn_apply(const float* __restrict__ X,
                                                const float* __restrict__ mu,
                                                const float* __restrict__ var,
                                                const float* __restrict__ g,
                                                const float* __restrict__ bta,
                                                _Float16* __restrict__ Y,
                                                int Nact, int total) {
  int t = blockIdx.x * 256 + threadIdx.x; if (t >= total) return;
  int n = t >> 8, c = t & 255;
  float r = 0.f;
  if (n < Nact) {
    float v = g[c] * (X[t] - mu[c]) * rsqrtf(var[c] + 1e-5f) + bta[c];
    r = v > 0.f ? v : 0.f;
  }
  Y[t] = (_Float16)r;
}

__global__ void __launch_bounds__(256) relu_copy(const float* __restrict__ X,
                                                 float* __restrict__ Y,
                                                 int Nact, int total) {
  int t = blockIdx.x * 256 + threadIdx.x; if (t >= total) return;
  int n = t >> 8;
  float v = (n < Nact) ? X[t] : 0.f;
  Y[t] = v > 0.f ? v : 0.f;
}

__global__ void __launch_bounds__(256) copy_to_hA(const float* __restrict__ H,
                                                  float* __restrict__ HA,
                                                  int colOff, int total) {
  int t = blockIdx.x * 256 + threadIdx.x; if (t >= total) return;
  int n = t >> 8, c = t & 255;
  HA[(size_t)n * 512 + colOff + c] = H[t];
}

// ------------------- A -> C2 scatter max (segment_max) ---------------------
__global__ void __launch_bounds__(256) segmax_scatter(const float* __restrict__ HA,
                                                      const int* __restrict__ dstg,
                                                      unsigned* __restrict__ enc, int total) {
  int t = blockIdx.x * 256 + threadIdx.x; if (t >= total) return;
  int n = t >> 9, c = t & 511;
  atomicMax(&enc[(size_t)dstg[n] * 512 + c], fenc(HA[t]));
}

__global__ void __launch_bounds__(256) hc_build(const unsigned* __restrict__ enc,
                                                const float* __restrict__ pca,
                                                float* __restrict__ xC, int total) {
  int t = blockIdx.x * 256 + threadIdx.x; if (t >= total) return;
  int n = t / 516, c = t - n * 516;
  float v = 0.f;
  if (n < NCA) v = (c < 512) ? fdec(enc[(size_t)n * 512 + c]) : pca[n * 4 + (c - 512)];
  xC[t] = v;
}

// ------------------------------ readout ------------------------------------
__global__ void __launch_bounds__(256) readout_cnt(const int* __restrict__ gid,
                                                   float* __restrict__ cnt, int N) {
  int t = blockIdx.x * 256 + threadIdx.x; if (t < N) atomicAdd(&cnt[gid[t]], 1.f);
}

// gid is sorted -> run-length-compress before atomics (block = 64 rows)
__global__ void __launch_bounds__(256) readout_sum(const float* __restrict__ H,
                                                   const int* __restrict__ gid,
                                                   float* __restrict__ sums,
                                                   int Nrows, int Cols) {
  int r0 = blockIdx.x * 64;
  int rend = r0 + 64; if (rend > Nrows) rend = Nrows;
  for (int c = threadIdx.x; c < Cols; c += 256) {
    float local = 0.f; int curg = -1;
    for (int r = r0; r < rend; r++) {
      int g = gid[r];
      if (g != curg) {
        if (curg >= 0) atomicAdd(&sums[curg * Cols + c], local);
        curg = g; local = 0.f;
      }
      local += H[(size_t)r * Cols + c];
    }
    if (curg >= 0) atomicAdd(&sums[curg * Cols + c], local);
  }
}

__global__ void __launch_bounds__(256) final_mlp(const float* __restrict__ sumA,
                                                 const float* __restrict__ sumC,
                                                 const float* __restrict__ cntA,
                                                 const float* __restrict__ cntC,
                                                 const float* __restrict__ W1,
                                                 const float* __restrict__ b1v,
                                                 const float* __restrict__ W2,
                                                 const float* __restrict__ b2v,
                                                 float* __restrict__ out) {
  __shared__ float sh[256];
  int t = threadIdx.x;
  for (int g = 0; g < GRP; g++) {
    float invA = 1.f / cntA[g], invC = 1.f / cntC[g];
    float acc = b1v[t];
    const float* sa = sumA + g * 512;
    const float* sc = sumC + g * 256;
    for (int k = 0; k < 512; k++) acc += sa[k] * invA * W1[k * 256 + t];
    for (int k = 0; k < 256; k++) acc += sc[k] * invC * W1[(512 + k) * 256 + t];
    float h = acc > 0.f ? acc : 0.f;
    sh[t] = h * W2[t];
    __syncthreads();
    for (int o = 128; o > 0; o >>= 1) { if (t < o) sh[t] += sh[t + o]; __syncthreads(); }
    if (t == 0) out[g] = sh[0] + b2v[0];
    __syncthreads();
  }
}

// ---------------------------------------------------------------------------
extern "C" void kernel_launch(void* const* d_in, const int* in_sizes, int n_in,
                              void* d_out, int out_size, void* d_ws, size_t ws_size,
                              hipStream_t stream) {
  (void)in_sizes; (void)out_size;
  if (n_in < 74) return;

  auto F32 = [&](int i) { return (const float*)d_in[i]; };
  const float* atom  = F32(0);
  const float* pca   = F32(1);
  const int* src_b1  = (const int*)d_in[2];
  const int* dst_b1  = (const int*)d_in[3];
  const int* src_b2  = (const int*)d_in[4];
  const int* dst_b2  = (const int*)d_in[5];
  const int* dst_g1  = (const int*)d_in[7];
  const int* src_i2  = (const int*)d_in[8];
  const int* dst_i2  = (const int*)d_in[9];
  const int* gid_A   = (const int*)d_in[10];
  const int* gid_C2  = (const int*)d_in[11];

  // ---- bump allocator over d_ws ----
  char* w = (char*)d_ws;
  auto alloc = [&](size_t bytes) -> void* {
    void* p = (void*)w;
    w += (bytes + 255) & ~(size_t)255;
    return p;
  };

  _Float16* x16    = (_Float16*)alloc((size_t)NA * 64 * 2);
  float*    bufF   = (float*)   alloc((size_t)NA * 256 * 4);
  _Float16* bufF16 = (_Float16*)alloc((size_t)NA * 256 * 2);
  float*    bufAcc = (float*)   alloc((size_t)NA * 256 * 4);
  float*    bufH   = (float*)   alloc((size_t)NA * 256 * 4);
  float*    bufHA  = (float*)   alloc((size_t)NA * 512 * 4);
  float*    el     = (float*)   alloc((size_t)NA * 4 * 4);
  float*    er     = (float*)   alloc((size_t)NA * 4 * 4);
  unsigned* mEnc   = (unsigned*)alloc((size_t)NA * 4 * 4);
  float*    ssum   = (float*)   alloc((size_t)NA * 4 * 4);
  float*    eE     = (float*)   alloc((size_t)EB * 4 * 4);
  float*    aE     = (float*)   alloc((size_t)EB * 4 * 4);
  float*    mu     = (float*)   alloc(256 * 4);
  float*    var    = (float*)   alloc(256 * 4);
  unsigned* hCEnc  = (unsigned*)alloc((size_t)NCP * 512 * 4);
  float*    xC     = (float*)   alloc((size_t)NCP * 516 * 4);
  float*    accC   = (float*)   alloc((size_t)NCP * 544 * 4);
  _Float16* xC16   = (_Float16*)alloc((size_t)NCP * 544 * 2);
  float*    bufFC  = (float*)   alloc((size_t)NCP * 256 * 4);
  _Float16* bufFC16= (_Float16*)alloc((size_t)NCP * 256 * 2);
  float*    hCc    = (float*)   alloc((size_t)NCP * 256 * 4);
  float*    sumA   = (float*)   alloc(32 * 512 * 4);
  float*    sumC   = (float*)   alloc(32 * 256 * 4);
  float*    cntA   = (float*)   alloc(32 * 4);
  float*    cntC   = (float*)   alloc(32 * 4);

  // transposed f16 weights
  struct WRec { int pidx; int K; int Kpad; };
  const WRec recs[16] = {
    {12, 64, 64},  {16, 256, 256}, {21, 256, 256}, {22, 256, 256}, {28, 256, 256}, {29, 256, 256},
    {34, 64, 64},  {38, 256, 256}, {43, 256, 256}, {44, 256, 256}, {50, 256, 256}, {51, 256, 256},
    {57, 516, 544},{58, 256, 256}, {64, 256, 256}, {65, 256, 256}
  };
  _Float16* wt[16];
  for (int i = 0; i < 16; i++) wt[i] = (_Float16*)alloc((size_t)256 * recs[i].Kpad * 2);

  if ((size_t)(w - (char*)d_ws) > ws_size) return;  // ~360 MB required

  auto B1 = [&](int n) { return dim3((unsigned)((n + 255) / 256)); };
  const dim3 TB(256);

  // ---- weight prep ----
  for (int i = 0; i < 16; i++) {
    int total = 256 * recs[i].Kpad;
    wt_prep<<<B1(total), TB, 0, stream>>>(F32(recs[i].pidx), wt[i],
                                          recs[i].K, 256, recs[i].Kpad, total);
  }

  // ---- positional encoding ----
  posenc_kernel<<<B1(NA), TB, 0, stream>>>(atom, x16, NA);

  // ---- GAT layer ----
  auto run_gat = [&](const _Float16* Ain, int K, const _Float16* Wt, const float* al,
                     const float* ar, const float* bvec, const int* src, const int* dst) {
    gemm_wmma<<<dim3(NAT), TB, 0, stream>>>(Ain, Wt, nullptr, bufF, K);
    gat_elr<<<B1(NA * 4), TB, 0, stream>>>(bufF, al, ar, el, er, NA * 4);
    fill_u32<<<B1(NA * 4), TB, 0, stream>>>(mEnc, ENC_NEGINF, NA * 4);
    fill_f32<<<B1(NA * 4), TB, 0, stream>>>(ssum, 0.f, NA * 4);
    fill_f32<<<B1(NA * 256), TB, 0, stream>>>(bufAcc, 0.f, NA * 256);
    gat_edge1<<<B1(EB * 4), TB, 0, stream>>>(el, er, src, dst, eE, mEnc, EB * 4);
    gat_edge2<<<B1(EB * 4), TB, 0, stream>>>(eE, dst, mEnc, aE, ssum, EB * 4);
    gat_edge3<<<B1(EB * 256), TB, 0, stream>>>(aE, bufF, src, dst, bufAcc, EB * 256);
    gat_fin<<<B1(NA * 256), TB, 0, stream>>>(bufAcc, ssum, bvec, bufH, bufF16, NA * 256);
  };

  // ---- GIN layer (atom graph) ----
  auto run_ginA = [&](const float* epsp, const _Float16* W1t, const float* b1v,
                      const _Float16* W2t, const float* b2v, const float* gv,
                      const float* btav, const int* src, const int* dst) {
    gin_init<<<B1(NA * 256), TB, 0, stream>>>(bufH, epsp, bufAcc, NA * 256);
    gin_scatter<<<B1(EB * 256), TB, 0, stream>>>(bufH, src, dst, bufAcc, 256, EB * 256);
    conv_pad_f16<<<B1(NA * 256), TB, 0, stream>>>(bufAcc, bufF16, 256, 256, NA * 256);
    gemm_wmma<<<dim3(NAT), TB, 0, stream>>>(bufF16, W1t, b1v, bufF, 256);
    bn_stats<<<dim3(256), TB, 0, stream>>>(bufF, mu, var, NA);
    bn_apply<<<B1(NA * 256), TB, 0, stream>>>(bufF, mu, var, gv, btav, bufF16, NA, NA * 256);
    gemm_wmma<<<dim3(NAT), TB, 0, stream>>>(bufF16, W2t, b2v, bufAcc, 256);
    relu_copy<<<B1(NA * 256), TB, 0, stream>>>(bufAcc, bufH, NA, NA * 256);
  };

  // ---- per edge-type pipeline ----
  const int* srcs[2] = { src_b1, src_b2 };
  const int* dsts[2] = { dst_b1, dst_b2 };
  const int  base[2] = { 12, 34 };
  for (int i = 0; i < 2; i++) {
    int Bp = base[i];
    const int* s = srcs[i]; const int* d = dsts[i];
    run_gat(x16,    64,  wt[i * 6 + 0], F32(Bp + 1), F32(Bp + 2), F32(Bp + 3), s, d);
    run_gat(bufF16, 256, wt[i * 6 + 1], F32(Bp + 5), F32(Bp + 6), F32(Bp + 7), s, d);
    run_ginA(F32(Bp + 8),  wt[i * 6 + 2], F32(Bp + 11), wt[i * 6 + 3], F32(Bp + 12),
             F32(Bp + 14), F32(Bp + 13), s, d);
    run_ginA(F32(Bp + 15), wt[i * 6 + 4], F32(Bp + 18), wt[i * 6 + 5], F32(Bp + 19),
             F32(Bp + 21), F32(Bp + 20), s, d);
    copy_to_hA<<<B1(NA * 256), TB, 0, stream>>>(bufH, bufHA, i * 256, NA * 256);
  }

  // ---- segment max A -> C2 and concat pca ----
  fill_u32<<<B1(NCP * 512), TB, 0, stream>>>(hCEnc, ENC_NEGINF, NCP * 512);
  segmax_scatter<<<B1(NA * 512), TB, 0, stream>>>(bufHA, dst_g1, hCEnc, NA * 512);
  hc_build<<<B1(NCP * 516), TB, 0, stream>>>(hCEnc, pca, xC, NCP * 516);

  // ---- GIN layers on C2 graph ----
  auto run_ginC = [&](const float* xin, int Din, int Kpad, const float* epsp,
                      const _Float16* W1t, const float* b1v, const _Float16* W2t,
                      const float* b2v, const float* gv, const float* btav, float* xout) {
    gin_init<<<B1(NCP * Din), TB, 0, stream>>>(xin, epsp, accC, NCP * Din);
    gin_scatter<<<B1(EI * Din), TB, 0, stream>>>(xin, src_i2, dst_i2, accC, Din, EI * Din);
    conv_pad_f16<<<B1(NCP * Kpad), TB, 0, stream>>>(accC, xC16, Din, Kpad, NCP * Kpad);
    gemm_wmma<<<dim3(NCT), TB, 0, stream>>>(xC16, W1t, b1v, bufFC, Kpad);
    bn_stats<<<dim3(256), TB, 0, stream>>>(bufFC, mu, var, NCA);
    bn_apply<<<B1(NCP * 256), TB, 0, stream>>>(bufFC, mu, var, gv, btav, bufFC16, NCA, NCP * 256);
    gemm_wmma<<<dim3(NCT), TB, 0, stream>>>(bufFC16, W2t, b2v, accC, 256);
    relu_copy<<<B1(NCP * 256), TB, 0, stream>>>(accC, xout, NCA, NCP * 256);
  };
  run_ginC(xC,  516, 544, F32(56), wt[12], F32(59), wt[13], F32(60), F32(62), F32(61), hCc);
  run_ginC(hCc, 256, 256, F32(63), wt[14], F32(66), wt[15], F32(67), F32(69), F32(68), hCc);

  // ---- readout + output MLP ----
  fill_f32<<<B1(32 * 512), TB, 0, stream>>>(sumA, 0.f, 32 * 512);
  fill_f32<<<B1(32 * 256), TB, 0, stream>>>(sumC, 0.f, 32 * 256);
  fill_f32<<<B1(32), TB, 0, stream>>>(cntA, 0.f, 32);
  fill_f32<<<B1(32), TB, 0, stream>>>(cntC, 0.f, 32);
  readout_cnt<<<B1(NA), TB, 0, stream>>>(gid_A, cntA, NA);
  readout_cnt<<<B1(NCA), TB, 0, stream>>>(gid_C2, cntC, NCA);
  readout_sum<<<dim3((NA + 63) / 64), TB, 0, stream>>>(bufHA, gid_A, sumA, NA, 512);
  readout_sum<<<dim3((NCA + 63) / 64), TB, 0, stream>>>(hCc, gid_C2, sumC, NCA, 256);
  final_mlp<<<dim3(1), TB, 0, stream>>>(sumA, sumC, cntA, cntC,
                                        F32(70), F32(72), F32(71), F32(73), (float*)d_out);
}
